// AlibiAttention_59167469469719
// MI455X (gfx1250) — matrix operand order repaired and checked by reading, not compiled, not run on hardware
//
#include <hip/hip_runtime.h>
#include <hip/hip_bf16.h>

#define HNUM 16
#define HD 64
#define NSEQ 2048
#define DMODEL 1024
#define BATCH 2
#define MTOT (BATCH * NSEQ) /* 4096 */

typedef __attribute__((ext_vector_type(16))) __bf16        v16bf;
typedef __attribute__((ext_vector_type(8)))  float         v8f;
typedef __attribute__((ext_vector_type(8)))  unsigned int  v8u;
typedef __attribute__((ext_vector_type(4)))  unsigned int  v4u;
typedef __attribute__((ext_vector_type(4)))  int           v4i;

static_assert(sizeof(v16bf) == 32, "v16bf must be 32 bytes");

// ---- async Global->LDS staging (CDNA5 path) with compile-safe fallback ----
#if defined(__has_builtin)
#  if __has_builtin(__builtin_amdgcn_global_load_async_to_lds_b128)
#    define USE_ASYNC_LDS 1
#  endif
#endif

typedef __attribute__((address_space(1))) v4i GV4;   // global int4
typedef __attribute__((address_space(3))) v4i LV4;   // LDS int4

__device__ __forceinline__ void stage16(const unsigned short* gsrc, unsigned short* ldst) {
#ifdef USE_ASYNC_LDS
  __builtin_amdgcn_global_load_async_to_lds_b128((GV4*)gsrc, (LV4*)ldst, 0, 0);
#else
  v4u t = *(const v4u*)gsrc;
  *(v4u*)ldst = t;
#endif
}

__device__ __forceinline__ void stage_wait() {
#ifdef USE_ASYNC_LDS
#  if __has_builtin(__builtin_amdgcn_s_wait_asynccnt)
  __builtin_amdgcn_s_wait_asynccnt(0);
#  else
  asm volatile("s_wait_asynccnt 0" ::: "memory");
#  endif
#endif
}

__device__ __forceinline__ unsigned short f2bf(float f) {
  unsigned int u = __float_as_uint(f);
  u += 0x7fffu + ((u >> 16) & 1u);   // round-to-nearest-even
  return (unsigned short)(u >> 16);
}

__device__ __forceinline__ v16bf load_bfrag(const unsigned short* p) {   // 32B-aligned
  return __builtin_bit_cast(v16bf, *(const v8u*)p);
}

__device__ __forceinline__ v16bf lds_bfrag(const unsigned short* p) {    // 16B-aligned
  union { v16bf v; v4u q[2]; } u;
  u.q[0] = *(const v4u*)p;
  u.q[1] = *(const v4u*)(p + 8);
  return u.v;
}

// element offset of pair p within a 16x32 16-bit A fragment row
__device__ __forceinline__ int a_pair_off(int p) {
  return 2 * p;
}

// ---------------------------------------------------------------------------
// Kernel 1: fp32 -> bf16 conversions; weights transposed so B-fragments are
// contiguous per-lane strips.
// ---------------------------------------------------------------------------
__global__ void convert_kernel(const float* __restrict__ x,
                               const float* __restrict__ wq, const float* __restrict__ wk,
                               const float* __restrict__ wv, const float* __restrict__ wo,
                               unsigned short* __restrict__ xb,
                               unsigned short* __restrict__ wqt, unsigned short* __restrict__ wkt,
                               unsigned short* __restrict__ wvt, unsigned short* __restrict__ wot) {
  int idx = blockIdx.x * blockDim.x + threadIdx.x;     // 0 .. 4M-1
  xb[idx] = f2bf(x[idx]);
  if (idx < DMODEL * DMODEL) {
    int k = idx >> 10;
    int j = idx & (DMODEL - 1);
    int t = j * DMODEL + k;                            // transpose: Wt[j][k] = W[k][j]
    wqt[t] = f2bf(wq[idx]);
    wkt[t] = f2bf(wk[idx]);
    wvt[t] = f2bf(wv[idx]);
    wot[t] = f2bf(wo[idx]);
  }
}

// ---------------------------------------------------------------------------
// Kernel 2: Q/K/V projections. Per-wave 32(M) x 64(N) tile, K in 32-chunks:
// 8 WMMAs per chunk, B-fragments reused across the two M sub-tiles.
// z=0 -> Q [b,H,n,HD], z=1 -> K [b,H,n,HD], z=2 -> V transposed [b,H,HD,n].
// ---------------------------------------------------------------------------
__global__ void __launch_bounds__(256) qkv_kernel(
    const unsigned short* __restrict__ xb,
    const unsigned short* __restrict__ wqt,
    const unsigned short* __restrict__ wkt,
    const unsigned short* __restrict__ wvt,
    unsigned short* __restrict__ Qb,
    unsigned short* __restrict__ Kb,
    unsigned short* __restrict__ Vtb) {
  const int lane = threadIdx.x & 31;
  const int g = lane >> 4;
  const int ln = lane & 15;
  const int mtile = blockIdx.x * 8 + (threadIdx.x >> 5);  // 0..127
  const int row0 = mtile * 32;
  const int N0 = blockIdx.y * 64;
  const int z = blockIdx.z;
  const unsigned short* wt = (z == 0) ? wqt : (z == 1) ? wkt : wvt;

  v8f acc[2][4] = {};
  const unsigned short* xrow0 = xb + (size_t)(row0 + ln) * DMODEL;
  const unsigned short* xrow1 = xb + (size_t)(row0 + 16 + ln) * DMODEL;
  for (int k0 = 0; k0 < DMODEL; k0 += 32) {
    union { v16bf v; unsigned int u[8]; } a[2];
#pragma unroll
    for (int p = 0; p < 8; ++p) {
      a[0].u[p] = *(const unsigned int*)(xrow0 + k0 + g * 16 + a_pair_off(p));
      a[1].u[p] = *(const unsigned int*)(xrow1 + k0 + g * 16 + a_pair_off(p));
    }
    const unsigned short* wbase = wt + k0 + g * 16;
#pragma unroll
    for (int t = 0; t < 4; ++t) {
      v16bf b = load_bfrag(wbase + (size_t)(N0 + t * 16 + ln) * DMODEL);
#pragma unroll
      for (int m = 0; m < 2; ++m)
        acc[m][t] = __builtin_amdgcn_wmma_f32_16x16x32_bf16(false, a[m].v, false, b,
                                                            (short)0, acc[m][t], false, false);
    }
  }

#pragma unroll
  for (int m = 0; m < 2; ++m) {
#pragma unroll
    for (int t = 0; t < 4; ++t) {
      int j = N0 + t * 16 + ln;
      int h = j >> 6, dd = j & 63;
#pragma unroll
      for (int r = 0; r < 8; ++r) {
        int i = row0 + m * 16 + r + 8 * g;        // global token row 0..4095
        int bi = i >> 11, tok = i & (NSEQ - 1);
        unsigned short val = f2bf(acc[m][t][r]);
        if (z == 0)      Qb [((size_t)(bi * HNUM + h) * NSEQ + tok) * HD + dd] = val;
        else if (z == 1) Kb [((size_t)(bi * HNUM + h) * NSEQ + tok) * HD + dd] = val;
        else             Vtb[((size_t)(bi * HNUM + h) * HD + dd) * NSEQ + tok] = val;
      }
    }
  }
}

// ---------------------------------------------------------------------------
// Kernel 3: cooperative flash attention with ALiBi + causal mask.
// One WG (8 waves) = one (b,h) and 128 consecutive queries (16 per wave).
// K/V 32-key blocks are staged into double-buffered LDS with async loads and
// shared by all 8 waves (8x L2-traffic reduction + copy/compute overlap).
// LDS rows padded (stride 72/40/40 elements) to avoid bank conflicts.
// ---------------------------------------------------------------------------
#define KL_STRIDE 72
#define VL_STRIDE 40
#define PL_STRIDE 40

__global__ void __launch_bounds__(256) attn_kernel(
    const unsigned short* __restrict__ Qb,
    const unsigned short* __restrict__ Kb,
    const unsigned short* __restrict__ Vtb,
    unsigned short* __restrict__ ctx) {
  __shared__ unsigned short Kl[2][32 * KL_STRIDE];   // [key][dd], padded
  __shared__ unsigned short Vl[2][64 * VL_STRIDE];   // [dd][key], padded
  __shared__ unsigned short Pl[8][16 * PL_STRIDE];   // per-wave P tile, padded

  const int tid = threadIdx.x;
  const int lane = tid & 31;
  const int g = lane >> 4, ln = lane & 15;
  const int wv = tid >> 5;                     // wave 0..7
  const int bh = blockIdx.x;                   // 0..31
  const int qsb = blockIdx.y * 128;            // query super-block start
  const int qb = qsb + wv * 16;                // this wave's 16 queries
  const int bi = bh >> 4, h = bh & 15;
  unsigned short* pb = Pl[wv];

  const unsigned short* Qp = Qb + (size_t)bh * NSEQ * HD;
  const unsigned short* Kp = Kb + (size_t)bh * NSEQ * HD;
  const unsigned short* Vp = Vtb + (size_t)bh * HD * NSEQ;

  const float slope = exp2f(-0.5f * (float)(h + 1));   // get_slopes(16)

  // Q A-fragments for the two 32-wide K chunks of HD=64
  union { v16bf v; unsigned int u[8]; } qa[2];
#pragma unroll
  for (int c = 0; c < 2; ++c) {
    const unsigned short* qr = Qp + (size_t)(qb + ln) * HD + c * 32 + g * 16;
#pragma unroll
    for (int p = 0; p < 8; ++p) qa[c].u[p] = *(const unsigned int*)(qr + a_pair_off(p));
  }

  float mrow[8], lrow[8];
#pragma unroll
  for (int r = 0; r < 8; ++r) { mrow[r] = -1e30f; lrow[r] = 0.f; }
  v8f acc[4] = {};

  // cooperative staging of one 32-key block: K (32x64) and V^T (64x32)
  const int krow = tid >> 3, kseg = tid & 7;   // 256 thr: K 32 rows x 8 segs of 16B
  const int vrow = tid >> 2, vseg = tid & 3;   // 256 thr: V 64 rows x 4 segs of 16B
  auto stage_block = [&](int kb, int buf) {
    stage16(Kp + (size_t)(kb + krow) * HD + kseg * 8,
            &Kl[buf][krow * KL_STRIDE + kseg * 8]);
    stage16(Vp + (size_t)vrow * NSEQ + kb + vseg * 8,
            &Vl[buf][vrow * VL_STRIDE + vseg * 8]);
  };

  const int nblocks = qsb / 32 + 4;            // covers keys 0 .. qsb+127
  stage_block(0, 0);

  for (int ib = 0; ib < nblocks; ++ib) {
    const int kb = ib * 32;
    const int cur = ib & 1;
    stage_wait();                              // my async loads into `cur` done
    __syncthreads();                           // everyone's `cur` data visible
    if (ib + 1 < nblocks) stage_block(kb + 32, cur ^ 1);

    if (kb <= qb + 15) {                       // wave-uniform skip of dead blocks
      // S = Q * K^T (two 16-key tiles, two 32-wide K chunks) from LDS
      v8f s[2] = {};
#pragma unroll
      for (int t = 0; t < 2; ++t) {
#pragma unroll
        for (int c = 0; c < 2; ++c) {
          v16bf bf = lds_bfrag(&Kl[cur][(t * 16 + ln) * KL_STRIDE + c * 32 + g * 16]);
          s[t] = __builtin_amdgcn_wmma_f32_16x16x32_bf16(false, qa[c].v, false, bf,
                                                         (short)0, s[t], false, false);
        }
      }

      // ALiBi + causal mask + online softmax (rows live in 16-lane groups)
#pragma unroll
      for (int r = 0; r < 8; ++r) {
        const int q = qb + r + 8 * g;
        const int kc0 = kb + ln, kc1 = kb + 16 + ln;
        float sv0 = (kc0 > q) ? -1e30f : (s[0][r] - slope * (float)(q - kc0)) * 0.125f;
        float sv1 = (kc1 > q) ? -1e30f : (s[1][r] - slope * (float)(q - kc1)) * 0.125f;
        float mnew = fmaxf(mrow[r], fmaxf(sv0, sv1));
#pragma unroll
        for (int off = 1; off < 16; off <<= 1)
          mnew = fmaxf(mnew, __shfl_xor(mnew, off, 32));
        const float corr = __expf(mrow[r] - mnew);
        mrow[r] = mnew;
        float p0 = __expf(sv0 - mnew);
        float p1 = __expf(sv1 - mnew);
        float ls = p0 + p1;
#pragma unroll
        for (int off = 1; off < 16; off <<= 1)
          ls += __shfl_xor(ls, off, 32);
        lrow[r] = lrow[r] * corr + ls;
#pragma unroll
        for (int t = 0; t < 4; ++t) acc[t][r] *= corr;
        pb[(r + 8 * g) * PL_STRIDE + ln] = f2bf(p0);
        pb[(r + 8 * g) * PL_STRIDE + 16 + ln] = f2bf(p1);
      }

      // same-wave LDS RAW fence, then read P back as an A-fragment
      asm volatile("s_wait_dscnt 0" ::: "memory");
      union { v16bf v; unsigned int u[8]; } pa;
      const unsigned short* pr = pb + ln * PL_STRIDE + g * 16;
#pragma unroll
      for (int p = 0; p < 8; ++p) pa.u[p] = *(const unsigned int*)(pr + a_pair_off(p));

      // ctx += P * V  (V^T rows in LDS: per-lane contiguous key strip)
#pragma unroll
      for (int t = 0; t < 4; ++t) {
        v16bf vf = lds_bfrag(&Vl[cur][(t * 16 + ln) * VL_STRIDE + g * 16]);
        acc[t] = __builtin_amdgcn_wmma_f32_16x16x32_bf16(false, pa.v, false, vf,
                                                         (short)0, acc[t], false, false);
      }
    }
    __syncthreads();                           // all reads of `cur` done before reuse
  }

  // normalize + store ctx as bf16 [b, n, H*HD]
#pragma unroll
  for (int t = 0; t < 4; ++t) {
    int dd = t * 16 + ln;
#pragma unroll
    for (int r = 0; r < 8; ++r) {
      int q = qb + r + 8 * g;
      ctx[(size_t)(bi * NSEQ + q) * DMODEL + h * HD + dd] = f2bf(acc[t][r] / lrow[r]);
    }
  }
}

// ---------------------------------------------------------------------------
// Kernel 4: out = ctx @ Wo + bo, fp32 result. 32x64 per-wave tile.
// ---------------------------------------------------------------------------
__global__ void __launch_bounds__(256) out_kernel(
    const unsigned short* __restrict__ cb,
    const unsigned short* __restrict__ wot,
    const float* __restrict__ bo,
    float* __restrict__ out) {
  const int lane = threadIdx.x & 31;
  const int g = lane >> 4, ln = lane & 15;
  const int mtile = blockIdx.x * 8 + (threadIdx.x >> 5);  // 0..127
  const int row0 = mtile * 32;
  const int N0 = blockIdx.y * 64;

  v8f acc[2][4] = {};
  const unsigned short* arow0 = cb + (size_t)(row0 + ln) * DMODEL;
  const unsigned short* arow1 = cb + (size_t)(row0 + 16 + ln) * DMODEL;
  for (int k0 = 0; k0 < DMODEL; k0 += 32) {
    union { v16bf v; unsigned int u[8]; } a[2];
#pragma unroll
    for (int p = 0; p < 8; ++p) {
      a[0].u[p] = *(const unsigned int*)(arow0 + k0 + g * 16 + a_pair_off(p));
      a[1].u[p] = *(const unsigned int*)(arow1 + k0 + g * 16 + a_pair_off(p));
    }
#pragma unroll
    for (int t = 0; t < 4; ++t) {
      v16bf b = load_bfrag(wot + (size_t)(N0 + t * 16 + ln) * DMODEL + k0 + g * 16);
#pragma unroll
      for (int m = 0; m < 2; ++m)
        acc[m][t] = __builtin_amdgcn_wmma_f32_16x16x32_bf16(false, a[m].v, false, b,
                                                            (short)0, acc[m][t], false, false);
    }
  }
#pragma unroll
  for (int m = 0; m < 2; ++m) {
#pragma unroll
    for (int t = 0; t < 4; ++t) {
      int j = N0 + t * 16 + ln;
      float bj = bo[j];
#pragma unroll
      for (int r = 0; r < 8; ++r)
        out[(size_t)(row0 + m * 16 + r + 8 * g) * DMODEL + j] = acc[m][t][r] + bj;
    }
  }
}

// ---------------------------------------------------------------------------
extern "C" void kernel_launch(void* const* d_in, const int* in_sizes, int n_in,
                              void* d_out, int out_size, void* d_ws, size_t ws_size,
                              hipStream_t stream) {
  (void)in_sizes; (void)n_in; (void)out_size; (void)ws_size;
  const float* x  = (const float*)d_in[0];
  const float* Wq = (const float*)d_in[1];
  const float* Wk = (const float*)d_in[2];
  const float* Wv = (const float*)d_in[3];
  const float* Wo = (const float*)d_in[4];
  const float* bo = (const float*)d_in[5];
  float* out = (float*)d_out;

  // workspace layout (bf16/ushort elements), total ~48 MB
  unsigned short* xb  = (unsigned short*)d_ws;                    // 4096*1024
  unsigned short* wqt = xb  + (size_t)MTOT * DMODEL;
  unsigned short* wkt = wqt + (size_t)DMODEL * DMODEL;
  unsigned short* wvt = wkt + (size_t)DMODEL * DMODEL;
  unsigned short* wot = wvt + (size_t)DMODEL * DMODEL;
  unsigned short* Qb  = wot + (size_t)DMODEL * DMODEL;            // [b,H,n,HD]
  unsigned short* Kb  = Qb  + (size_t)MTOT * DMODEL;              // [b,H,n,HD]
  unsigned short* Vtb = Kb  + (size_t)MTOT * DMODEL;              // [b,H,HD,n]
  unsigned short* cb  = Vtb + (size_t)MTOT * DMODEL;              // [b,n,d]

  convert_kernel<<<(MTOT * DMODEL) / 256, 256, 0, stream>>>(
      x, Wq, Wk, Wv, Wo, xb, wqt, wkt, wvt, wot);
  qkv_kernel<<<dim3(16, 16, 3), 256, 0, stream>>>(xb, wqt, wkt, wvt, Qb, Kb, Vtb);
  attn_kernel<<<dim3(32, 16), 256, 0, stream>>>(Qb, Kb, Vtb, cb);
  out_kernel<<<dim3(16, 16), 256, 0, stream>>>(cb, wot, bo, out);
}
